// Decoder_35330400977229
// MI455X (gfx1250) — compile-verified, matrix operand order checked by reading, not run on hardware
//
#include <hip/hip_runtime.h>
#include <hip/hip_bf16.h>

// ---------------------------------------------------------------------------
// CDNA5 WMMA types (wave32): v_wmma_f32_16x16x32_bf16
// ---------------------------------------------------------------------------
typedef __attribute__((ext_vector_type(16))) __bf16 bf16x16;
typedef __attribute__((ext_vector_type(8)))  float  f32x8;

__device__ __forceinline__ unsigned short bf16bits(float x) {
    union { float f; unsigned u; } a; a.f = x;
    return (unsigned short)((a.u + 0x7FFFu + ((a.u >> 16) & 1u)) >> 16);  // RNE
}
__device__ __forceinline__ unsigned pack2bf(float x0, float x1) {
    return (unsigned)bf16bits(x0) | ((unsigned)bf16bits(x1) << 16);
}
__device__ __forceinline__ __bf16 f2bf(float x) {
    unsigned short s = bf16bits(x);
    __bf16 b; __builtin_memcpy(&b, &s, 2); return b;
}

__device__ __forceinline__ f32x8 wmma_bf16(bf16x16 a, bf16x16 b, f32x8 c) {
    return __builtin_amdgcn_wmma_f32_16x16x32_bf16(
        false, a, false, b, (short)0, c, false, false);
}

// Swizzle helpers: CDNA5 16-bit operand layout for a 16x32 (A) / 32x16 (B) tile.
// lane<16 holds K {0..7}U{16..23}, lane>=16 holds K {8..15}U{24..31};
// element index e within the 16 bf16 of a lane: e = (k&7) + ((k>>4)<<3).
// A: lane = (m&15) + 16*((k>>3)&1) ; B: lane = (n&15) + 16*((k>>3)&1)
__device__ __forceinline__ int frag_lane(int mn, int k) { return (mn & 15) + 16 * ((k >> 3) & 1); }
__device__ __forceinline__ int frag_elem(int k)         { return (k & 7) + ((k >> 4) << 3); }

// ---------------------------------------------------------------------------
// GEMM: C[M,N] = A[M,K] @ B[K,N] (+bias)(relu?).  Block 256 thr = 8 waves
// (2 M x 4 N). Block tile 32x256, wave tile 16x64 -> 4 WMMA / K-step / wave.
// LDS tiles stored pre-swizzled as bf16 fragments; frag load = 32B vector read.
// ---------------------------------------------------------------------------
#define BLK_M 32
#define BLK_N 256
#define BLK_K 32

__global__ __launch_bounds__(256) void gemm_bias_wmma(
    const float* __restrict__ A, const float* __restrict__ Bm,
    const float* __restrict__ bias, float* __restrict__ C,
    int M, int K, int N, int relu)
{
    __shared__ alignas(32) unsigned short sA[2 * 32 * 16];    // 2 M-frag groups
    __shared__ alignas(32) unsigned short sB[16 * 32 * 16];   // 16 N-col groups
    unsigned* sAu = (unsigned*)sA;
    unsigned* sBu = (unsigned*)sB;

    const int tid  = threadIdx.x;
    const int wave = tid >> 5, lane = tid & 31;
    const int wm = wave >> 2, wn = wave & 3;
    const int m0 = blockIdx.x * BLK_M;
    const int n0 = blockIdx.y * BLK_N;

    f32x8 acc[4] = {};

    for (int k0 = 0; k0 < K; k0 += BLK_K) {
        // stage A: 32x32 f32 -> swizzled bf16 (512 packed pairs along k)
        for (int p = tid; p < 512; p += 256) {
            int m = p >> 4, k = (p & 15) * 2;
            float2 v = *(const float2*)&A[(size_t)(m0 + m) * K + (k0 + k)];
            sAu[(((m >> 4) * 32 + frag_lane(m, k)) * 16 + frag_elem(k)) >> 1] =
                pack2bf(v.x, v.y);
        }
        // stage B: 32x256 f32 -> swizzled bf16 (4096 pairs along k, strided)
        for (int p = tid; p < 4096; p += 256) {
            int n = p & 255, k = (p >> 8) * 2;
            float x0 = Bm[(size_t)(k0 + k) * N + (n0 + n)];
            float x1 = Bm[(size_t)(k0 + k + 1) * N + (n0 + n)];
            sBu[(((n >> 4) * 32 + frag_lane(n, k)) * 16 + frag_elem(k)) >> 1] =
                pack2bf(x0, x1);
        }
        __syncthreads();

        bf16x16 af = *(const bf16x16*)&sA[(wm * 32 + lane) * 16];
#pragma unroll
        for (int t = 0; t < 4; ++t) {
            bf16x16 bf = *(const bf16x16*)&sB[((wn * 4 + t) * 32 + lane) * 16];
            acc[t] = wmma_bf16(af, bf, acc[t]);
        }
        __syncthreads();
    }

    const int hf = lane >> 4, col = lane & 15;
#pragma unroll
    for (int t = 0; t < 4; ++t) {
        int nc = n0 + (wn * 4 + t) * 16 + col;
        float bv = bias ? bias[nc] : 0.0f;
#pragma unroll
        for (int r = 0; r < 8; ++r) {
            float v = acc[t][r] + bv;
            if (relu) v = fmaxf(v, 0.0f);
            C[(size_t)(m0 + wm * 16 + r + 8 * hf) * N + nc] = v;
        }
    }
}

// ---------------------------------------------------------------------------
// Flash attention, d_h = 32. Block = 256 thr = 8 waves; wave owns a 16-row
// q-tile; the block shares staged K^T / V fragments per 32-key chunk.
// Per chunk per wave: 2 score WMMAs + online softmax + 2 PV WMMAs.
// ---------------------------------------------------------------------------
#define DH 32

__global__ __launch_bounds__(256) void attn_flash_wmma(
    const float* __restrict__ Q, const float* __restrict__ Km,
    const float* __restrict__ Vm, float* __restrict__ O,
    int A, int D)
{
    __shared__ alignas(32) unsigned short sK[2 * 32 * 16];   // K^T frags (2 key grps)
    __shared__ alignas(32) unsigned short sV[2 * 32 * 16];   // V frags (2 d grps)
    __shared__ alignas(32) unsigned short sP[8 * 32 * 16];   // per-wave P frags
    unsigned* sKu = (unsigned*)sK;
    unsigned* sVu = (unsigned*)sV;

    const int tid  = threadIdx.x;
    const int wave = tid >> 5, lane = tid & 31;
    const int h = blockIdx.y, b = blockIdx.z;
    const int q0 = (blockIdx.x * 8 + wave) * 16;
    const int hoff = h * DH;
    const int hf = lane >> 4, col = lane & 15;
    const float scale = 0.17677669529663687f;   // 1/sqrt(32)

    // Q fragment, scale folded in (gathered once from global)
    bf16x16 aq;
    {
        const float* qbase = Q + ((size_t)b * A + q0) * D + hoff;
        int row = lane & 15, kb = hf * 8;
#pragma unroll
        for (int v = 0; v < 8; ++v) {
            int k = kb + ((v < 4) ? (2 * v) : (2 * v + 8));
            aq[2 * v]     = f2bf(qbase[row * D + k] * scale);
            aq[2 * v + 1] = f2bf(qbase[row * D + k + 1] * scale);
        }
    }

    float mrow[8], lrow[8];
#pragma unroll
    for (int r = 0; r < 8; ++r) { mrow[r] = -1e30f; lrow[r] = 0.0f; }
    f32x8 o0 = {}, o1 = {};
    unsigned short* sPw = sP + wave * 512;

    for (int ch = 0; ch < A / 32; ++ch) {
        __syncthreads();
        const size_t kvrow = (size_t)b * A + ch * 32;
        // stage K^T frags: logical B[k=d][n=key]; pairs contiguous along d
        for (int p = tid; p < 512; p += 256) {
            int key = p >> 4, d = (p & 15) * 2;
            float2 v = *(const float2*)&Km[(kvrow + key) * D + hoff + d];
            sKu[(((key >> 4) * 32 + frag_lane(key, d)) * 16 + frag_elem(d)) >> 1] =
                pack2bf(v.x, v.y);
        }
        // stage V frags: logical B[k=key][n=d]; pairs along key (strided)
        for (int p = tid; p < 512; p += 256) {
            int d = p & 31, key = (p >> 5) * 2;
            float x0 = Vm[(kvrow + key) * D + hoff + d];
            float x1 = Vm[(kvrow + key + 1) * D + hoff + d];
            sVu[(((d >> 4) * 32 + frag_lane(d, key)) * 16 + frag_elem(key)) >> 1] =
                pack2bf(x0, x1);
        }
        __syncthreads();

        bf16x16 bk0 = *(const bf16x16*)&sK[(0 * 32 + lane) * 16];
        bf16x16 bk1 = *(const bf16x16*)&sK[(1 * 32 + lane) * 16];
        f32x8 z = {};
        f32x8 s0 = wmma_bf16(aq, bk0, z);
        f32x8 s1 = wmma_bf16(aq, bk1, z);

        // online softmax per output row (rows split across half-waves)
        float p0[8], p1[8];
#pragma unroll
        for (int r = 0; r < 8; ++r) {
            float t = fmaxf(s0[r], s1[r]);
#pragma unroll
            for (int off = 8; off; off >>= 1) t = fmaxf(t, __shfl_xor(t, off, 32));
            float mnew = fmaxf(mrow[r], t);
            float sc = __expf(mrow[r] - mnew);
            mrow[r] = mnew;
            p0[r] = __expf(s0[r] - mnew);
            p1[r] = __expf(s1[r] - mnew);
            float su = p0[r] + p1[r];
#pragma unroll
            for (int off = 8; off; off >>= 1) su += __shfl_xor(su, off, 32);
            lrow[r] = lrow[r] * sc + su;
            o0[r] *= sc;
            o1[r] *= sc;
        }
        // scatter P (D-layout regs) into this wave's swizzled A-frag region
#pragma unroll
        for (int r = 0; r < 8; ++r) {
            int m = r + 8 * hf;
            int k0a = col;          // key tile 0
            int k1a = 16 + col;     // key tile 1
            sPw[frag_lane(m, k0a) * 16 + frag_elem(k0a)] = bf16bits(p0[r]);
            sPw[frag_lane(m, k1a) * 16 + frag_elem(k1a)] = bf16bits(p1[r]);
        }
        bf16x16 ap  = *(const bf16x16*)&sPw[lane * 16];
        bf16x16 bv0 = *(const bf16x16*)&sV[(0 * 32 + lane) * 16];
        bf16x16 bv1 = *(const bf16x16*)&sV[(1 * 32 + lane) * 16];
        o0 = wmma_bf16(ap, bv0, o0);
        o1 = wmma_bf16(ap, bv1, o1);
    }

    float* obase = O + ((size_t)b * A + q0) * D + hoff;
#pragma unroll
    for (int r = 0; r < 8; ++r) {
        float inv = 1.0f / lrow[r];
        obase[(size_t)(r + 8 * hf) * D + col]      = o0[r] * inv;
        obase[(size_t)(r + 8 * hf) * D + 16 + col] = o1[r] * inv;
    }
}

// ---------------------------------------------------------------------------
// out = LayerNorm(X + Y) * g + b  (one 256-wide row per block)
// ---------------------------------------------------------------------------
__global__ __launch_bounds__(256) void add_ln_kernel(
    const float* __restrict__ X, const float* __restrict__ Y,
    const float* __restrict__ g, const float* __restrict__ bta,
    float* __restrict__ O, int D)
{
    __shared__ float red[256];
    const size_t row = blockIdx.x;
    const int d = threadIdx.x;
    float v = X[row * D + d] + Y[row * D + d];

    red[d] = v; __syncthreads();
    for (int s = 128; s; s >>= 1) { if (d < s) red[d] += red[d + s]; __syncthreads(); }
    float mean = red[0] / (float)D;
    __syncthreads();

    float c = v - mean;
    red[d] = c * c; __syncthreads();
    for (int s = 128; s; s >>= 1) { if (d < s) red[d] += red[d + s]; __syncthreads(); }
    float var = red[0] / (float)D;

    O[row * D + d] = c * rsqrtf(var + 1e-5f) * g[d] + bta[d];
}

// ---------------------------------------------------------------------------
// norm + softmax weighted pooling: one block per batch (256 thr, D=256)
// ---------------------------------------------------------------------------
__global__ __launch_bounds__(256) void pool_kernel(
    const float* __restrict__ H, float* __restrict__ P, int A, int D)
{
    __shared__ float wts[512];
    const int b = blockIdx.x;
    const int tid = threadIdx.x, wave = tid >> 5, lane = tid & 31;
    const float* hb = H + (size_t)b * A * D;

    for (int a = wave; a < A; a += 8) {
        float s = 0.0f;
        for (int d = lane; d < D; d += 32) { float x = hb[(size_t)a * D + d]; s += x * x; }
#pragma unroll
        for (int off = 16; off; off >>= 1) s += __shfl_xor(s, off, 32);
        if (lane == 0) wts[a] = sqrtf(s);
    }
    __syncthreads();

    if (wave == 0) {
        float mx = -1e30f;
        for (int a = lane; a < A; a += 32) mx = fmaxf(mx, wts[a]);
#pragma unroll
        for (int off = 16; off; off >>= 1) mx = fmaxf(mx, __shfl_xor(mx, off, 32));
        float sum = 0.0f;
        for (int a = lane; a < A; a += 32) sum += __expf(wts[a] - mx);
#pragma unroll
        for (int off = 16; off; off >>= 1) sum += __shfl_xor(sum, off, 32);
        float inv = 1.0f / sum;
        for (int a = lane; a < A; a += 32) wts[a] = __expf(wts[a] - mx) * inv;
    }
    __syncthreads();

    const int d = tid;
    float acc = 0.0f;
    for (int a = 0; a < A; ++a) acc += wts[a] * hb[(size_t)a * D + d];
    P[(size_t)b * D + d] = acc;
}

// ---------------------------------------------------------------------------
// final tiny layer: out[B,2] = H[B,F] @ W[F,2] + b
// ---------------------------------------------------------------------------
__global__ void fc_out_kernel(const float* __restrict__ H, const float* __restrict__ W,
                              const float* __restrict__ bo, float* __restrict__ O,
                              int Bb, int F)
{
    int i = blockIdx.x * blockDim.x + threadIdx.x;
    if (i >= Bb * 2) return;
    int b = i >> 1, j = i & 1;
    float s = bo[j];
    for (int f = 0; f < F; ++f) s += H[(size_t)b * F + f] * W[f * 2 + j];
    O[i] = s;
}

// ---------------------------------------------------------------------------
// Launch
// ---------------------------------------------------------------------------
extern "C" void kernel_launch(void* const* d_in, const int* in_sizes, int n_in,
                              void* d_out, int out_size, void* d_ws, size_t ws_size,
                              hipStream_t stream) {
    const int Bb = 64, A = 512, D = 256, F = 512;
    const size_t BA = (size_t)Bb * A;      // 32768 rows
    const size_t S  = BA * D;              // one activation tensor (f32 elems)
    const int DD = D * D;

    const float* trg   = (const float*)d_in[0];
    const float* src   = (const float*)d_in[1];
    const float* W_sa  = (const float*)d_in[2];
    const float* b_sa  = (const float*)d_in[3];
    const float* W_ea  = (const float*)d_in[4];
    const float* b_ea  = (const float*)d_in[5];
    const float* ln_g  = (const float*)d_in[6];
    const float* ln_b  = (const float*)d_in[7];
    const float* W_fc1 = (const float*)d_in[8];
    const float* b_fc1 = (const float*)d_in[9];
    const float* W_fc2 = (const float*)d_in[10];
    const float* b_fc2 = (const float*)d_in[11];
    const float* W_out = (const float*)d_in[12];
    const float* b_out = (const float*)d_in[13];
    float* out = (float*)d_out;

    float* ws = (float*)d_ws;
    float* Qb = ws;           // 5 big buffers of S floats (~168 MB total)
    float* Kb = ws + S;
    float* Vb = ws + 2 * S;
    float* Tb = ws + 3 * S;
    float* Ub = ws + 4 * S;
    float* pooled = ws + 5 * S;                 // 64*256
    float* h1 = pooled + (size_t)Bb * D;        // 64*512
    float* h2 = h1 + (size_t)Bb * F;            // 64*512

    dim3 blk(256);
    dim3 gProj((int)(BA / BLK_M), D / BLK_N);   // 1024 x 1
    dim3 gAttn(A / 128, 8, Bb);                 // 4 x 8 x 64
    dim3 gLN((int)BA);

    // ---- self-attention ----
    gemm_bias_wmma<<<gProj, blk, 0, stream>>>(trg, W_sa + 0 * DD, b_sa + 0 * D, Qb, (int)BA, D, D, 0);
    gemm_bias_wmma<<<gProj, blk, 0, stream>>>(trg, W_sa + 1 * DD, b_sa + 1 * D, Kb, (int)BA, D, D, 0);
    gemm_bias_wmma<<<gProj, blk, 0, stream>>>(trg, W_sa + 2 * DD, b_sa + 2 * D, Vb, (int)BA, D, D, 0);
    attn_flash_wmma<<<gAttn, blk, 0, stream>>>(Qb, Kb, Vb, Tb, A, D);
    gemm_bias_wmma<<<gProj, blk, 0, stream>>>(Tb, W_sa + 3 * DD, b_sa + 3 * D, Qb, (int)BA, D, D, 0);
    add_ln_kernel<<<gLN, blk, 0, stream>>>(trg, Qb, ln_g, ln_b, Kb, D);   // h -> Kb

    // ---- cross-attention ----
    gemm_bias_wmma<<<gProj, blk, 0, stream>>>(Kb,  W_ea + 0 * DD, b_ea + 0 * D, Qb, (int)BA, D, D, 0);
    gemm_bias_wmma<<<gProj, blk, 0, stream>>>(src, W_ea + 1 * DD, b_ea + 1 * D, Vb, (int)BA, D, D, 0);
    gemm_bias_wmma<<<gProj, blk, 0, stream>>>(src, W_ea + 2 * DD, b_ea + 2 * D, Tb, (int)BA, D, D, 0);
    attn_flash_wmma<<<gAttn, blk, 0, stream>>>(Qb, Vb, Tb, Ub, A, D);
    gemm_bias_wmma<<<gProj, blk, 0, stream>>>(Ub, W_ea + 3 * DD, b_ea + 3 * D, Qb, (int)BA, D, D, 0);
    add_ln_kernel<<<gLN, blk, 0, stream>>>(Kb, Qb, ln_g, ln_b, Vb, D);    // h2 -> Vb

    // ---- pooling + FC stack ----
    pool_kernel<<<dim3(Bb), blk, 0, stream>>>(Vb, pooled, A, D);
    gemm_bias_wmma<<<dim3(Bb / BLK_M, F / BLK_N), blk, 0, stream>>>(pooled, W_fc1, b_fc1, h1, Bb, D, F, 1);
    gemm_bias_wmma<<<dim3(Bb / BLK_M, F / BLK_N), blk, 0, stream>>>(h1,     W_fc2, b_fc2, h2, Bb, F, F, 1);
    fc_out_kernel<<<dim3(1), dim3(128), 0, stream>>>(h2, W_out, b_out, out, Bb, F);
}
